// MambaModel_45561013076511
// MI455X (gfx1250) — matrix operand order repaired and checked by reading, not compiled, not run on hardware
//
#include <hip/hip_runtime.h>

typedef __attribute__((ext_vector_type(16))) __bf16 v16bf;
typedef __attribute__((ext_vector_type(8)))  float  v8f;

#define BATCH 128
#define SEQ   2048
#define FEAT  32
#define DM    64
#define DI    128
#define DSTATE 16
#define TT    16
#define NTILES (SEQ / TT)

// A-matrix (16x32 bf16) per-lane K index: VGPRs 0-3 = K0-7 / K8-15 (lane<16 / lane>=16),
// VGPRs 4-7 = K16-23 / K24-31  (ISA 7.12.2)
__device__ __forceinline__ int kA(int e, int lane) {
  return ((e >> 3) << 4) + ((lane >> 4) << 3) + (e & 7);
}
// B-matrix (32x16 bf16) per-lane K index: same as A: e0-7 = K0-7 / K8-15, e8-15 = K16-23 / K24-31
__device__ __forceinline__ int kB(int e, int lane) {
  return ((e >> 3) << 4) + ((lane >> 4) << 3) + (e & 7);
}

__device__ __forceinline__ v8f wmma_bf16(v16bf a, v16bf b, v8f c) {
  // 8 args: (neg_a, A, neg_b, B, c_mod, C, reuse_a, reuse_b)
  return __builtin_amdgcn_wmma_f32_16x16x32_bf16(false, a, false, b, (short)0, c,
                                                 false, false);
}

__global__ __launch_bounds__(256)
void mamba_fused_kernel(const float* __restrict__ xg,
                        const float* __restrict__ emb_w,
                        const float* __restrict__ emb_b,
                        const float* __restrict__ in_proj_w,
                        const float* __restrict__ conv_w,
                        const float* __restrict__ conv_b,
                        const float* __restrict__ x_proj_w,
                        const float* __restrict__ dt_proj_w,
                        const float* __restrict__ dt_proj_b,
                        const float* __restrict__ A_log,
                        const float* __restrict__ D_param,
                        const float* __restrict__ out_proj_w,
                        const float* __restrict__ fc_w,
                        const float* __restrict__ fc_b,
                        float* __restrict__ outp)
{
  // ---- weight fragments (bf16, fragment-linear: frag*512 + lane*16 + e) ----
  __shared__ __align__(32) __bf16 wemb[4  * 512];   // emb:     N=64  -> 4 n-tiles, K=32
  __shared__ __align__(32) __bf16 winp[32 * 512];   // in_proj: N=256 -> 16 n-tiles x 2 k-slices
  __shared__ __align__(32) __bf16 wxp [12 * 512];   // x_proj:  N=48  -> 3 n-tiles x 4 k-slices
  // ---- activations ----
  __shared__ float  xt  [TT * FEAT];                // x tile (f32)
  __shared__ __align__(32) __bf16 hbf[TT * DM];     // h tile (bf16)
  __shared__ float  xz  [TT * 256];                 // u | z  (f32)
  __shared__ float  us  [TT * DI];                  // silu(conv(u))
  __shared__ float  xdbl[TT * 48];                  // dt(4) | B(16) | C(16) | pad
  __shared__ float  dlt [TT * DI];                  // softplus delta
  __shared__ float  hist[3 * DI];                   // raw-u conv history (t-3..t-1)
  __shared__ float  yred[2 * DI];
  __shared__ float  yvec[DI];
  __shared__ float  outv[DM];

  const int tid  = threadIdx.x;
  const int lane = tid & 31;
  const int wave = tid >> 5;
  const int b    = blockIdx.x;

  // ================= one-time per-block weight prep =================
  for (int i = tid; i < 3 * DI; i += 256) hist[i] = 0.f;

  for (int idx = tid; idx < 4 * 512; idx += 256) {         // emb B frags
    int nt = idx >> 9, r = idx & 511, l = r >> 4, e = r & 15;
    int k = kB(e, l), n = nt * 16 + (l & 15);
    wemb[idx] = (__bf16)emb_w[n * FEAT + k];               // B[k][n] = emb_w[n][k]
  }
  for (int idx = tid; idx < 32 * 512; idx += 256) {        // in_proj B frags
    int f = idx >> 9, r = idx & 511, l = r >> 4, e = r & 15;
    int nt = f >> 1, ks = f & 1;
    int k = ks * 32 + kB(e, l), n = nt * 16 + (l & 15);
    winp[idx] = (__bf16)in_proj_w[n * DM + k];
  }
  for (int idx = tid; idx < 12 * 512; idx += 256) {        // x_proj B frags (pad N to 48)
    int f = idx >> 9, r = idx & 511, l = r >> 4, e = r & 15;
    int nt = f >> 2, ks = f & 3;
    int k = ks * 32 + kB(e, l), n = nt * 16 + (l & 15);
    wxp[idx] = (n < 36) ? (__bf16)x_proj_w[n * DI + k] : (__bf16)0.f;
  }

  // ---- per-thread persistent state: channel d, state half nh ----
  const int d  = tid & 127;
  const int nh = tid >> 7;
  float Arow[8];
  #pragma unroll
  for (int j = 0; j < 8; ++j) Arow[j] = -__expf(A_log[d * DSTATE + nh * 8 + j]);
  float st[8];
  #pragma unroll
  for (int j = 0; j < 8; ++j) st[j] = 0.f;

  const float cw0 = conv_w[d * 4 + 0], cw1 = conv_w[d * 4 + 1];
  const float cw2 = conv_w[d * 4 + 2], cw3 = conv_w[d * 4 + 3];
  const float cbias = conv_b[d];
  const float dw0 = dt_proj_w[d * 4 + 0], dw1 = dt_proj_w[d * 4 + 1];
  const float dw2 = dt_proj_w[d * 4 + 2], dw3 = dt_proj_w[d * 4 + 3];
  const float dtb = dt_proj_b[d];

  const float* xb = xg + (size_t)b * SEQ * FEAT;
  __syncthreads();

  // ================= sequence loop: 128 tiles of 16 timesteps =================
  for (int tile = 0; tile < NTILES; ++tile) {
    const bool last = (tile == NTILES - 1);

    // ---- S0: stream x tile into LDS; prefetch next tile ----
    for (int i = tid; i < TT * FEAT; i += 256) xt[i] = xb[tile * TT * FEAT + i];
    if (tile + 1 < NTILES)
      __builtin_prefetch(xb + (tile + 1) * TT * FEAT + tid * 2, 0, 0);
    __syncthreads();

    // ---- S1: embedding  [16x32] @ [32x64]  (waves 0-3, 1 WMMA each) ----
    if (wave < 4) {
      v16bf a;
      #pragma unroll
      for (int e = 0; e < 16; ++e)
        a[e] = (__bf16)xt[(lane & 15) * FEAT + kA(e, lane)];
      const v16bf* wp = (const v16bf*)&wemb[wave * 512 + lane * 16];
      v8f c = {};
      c = wmma_bf16(a, *wp, c);
      int n = wave * 16 + (lane & 15);
      float bias = emb_b[n];
      #pragma unroll
      for (int v = 0; v < 8; ++v) {
        int m = v + ((lane >> 4) << 3);
        hbf[m * DM + n] = (__bf16)(c[v] + bias);
      }
    }
    __syncthreads();

    // ---- S2: in_proj [16x64] @ [64x256]; z half only on final tile ----
    {
      v16bf a0, a1;
      #pragma unroll
      for (int e = 0; e < 16; ++e) {
        int kk = kA(e, lane);
        a0[e] = hbf[(lane & 15) * DM + kk];
        a1[e] = hbf[(lane & 15) * DM + 32 + kk];
      }
      int nmax = last ? 16 : 8;          // only u-half except at the last token tile
      for (int nt = wave; nt < nmax; nt += 8) {
        const v16bf* w0 = (const v16bf*)&winp[(nt * 2 + 0) * 512 + lane * 16];
        const v16bf* w1 = (const v16bf*)&winp[(nt * 2 + 1) * 512 + lane * 16];
        v8f c = {};
        c = wmma_bf16(a0, *w0, c);
        c = wmma_bf16(a1, *w1, c);
        int n = nt * 16 + (lane & 15);
        #pragma unroll
        for (int v = 0; v < 8; ++v) {
          int m = v + ((lane >> 4) << 3);
          xz[m * 256 + n] = c[v];
        }
      }
    }
    __syncthreads();

    // ---- S3: causal depthwise conv(4) + SiLU (VALU, 8 t-rows per thread) ----
    {
      int tb = nh * 8;
      #pragma unroll
      for (int q = 0; q < 8; ++q) {
        int t = tb + q;
        float u3 = xz[t * 256 + d];
        float u2 = (t >= 1) ? xz[(t - 1) * 256 + d] : hist[(2 + t) * DI + d];
        float u1 = (t >= 2) ? xz[(t - 2) * 256 + d] : hist[(1 + t) * DI + d];
        float u0 = (t >= 3) ? xz[(t - 3) * 256 + d] : hist[(0 + t) * DI + d];
        float acc = cbias + cw0 * u0 + cw1 * u1 + cw2 * u2 + cw3 * u3;
        us[t * DI + d] = acc / (1.f + __expf(-acc));       // silu
      }
    }
    __syncthreads();
    for (int i = tid; i < 3 * DI; i += 256)                // carry raw u history
      hist[i] = xz[(13 + (i >> 7)) * 256 + (i & 127)];
    __syncthreads();

    // ---- S4: x_proj [16x128] @ [128x48]  (waves 0-2, 4 WMMA each) ----
    if (wave < 3) {
      v8f c = {};
      #pragma unroll
      for (int ks = 0; ks < 4; ++ks) {
        v16bf a;
        #pragma unroll
        for (int e = 0; e < 16; ++e)
          a[e] = (__bf16)us[(lane & 15) * DI + ks * 32 + kA(e, lane)];
        const v16bf* wp = (const v16bf*)&wxp[(wave * 4 + ks) * 512 + lane * 16];
        c = wmma_bf16(a, *wp, c);
      }
      int n = wave * 16 + (lane & 15);
      #pragma unroll
      for (int v = 0; v < 8; ++v) {
        int m = v + ((lane >> 4) << 3);
        xdbl[m * 48 + n] = c[v];
      }
    }
    __syncthreads();

    // ---- S5: dt_proj (rank 4) + softplus ----
    {
      int tb = nh * 8;
      #pragma unroll
      for (int q = 0; q < 8; ++q) {
        int t = tb + q;
        float pre = dtb + xdbl[t * 48 + 0] * dw0 + xdbl[t * 48 + 1] * dw1
                        + xdbl[t * 48 + 2] * dw2 + xdbl[t * 48 + 3] * dw3;
        float sp = (pre > 15.f) ? pre : __logf(1.f + __expf(pre));
        dlt[t * DI + d] = sp;
      }
    }
    __syncthreads();

    // ---- S6: selective scan, 16 sequential steps, state in VGPRs ----
    for (int t = 0; t < TT; ++t) {
      float dl = dlt[t * DI + d];
      float uu = us [t * DI + d];
      float du = dl * uu;
      #pragma unroll
      for (int j = 0; j < 8; ++j) {
        float bj = xdbl[t * 48 + 4 + nh * 8 + j];          // B_t[n]
        st[j] = __expf(dl * Arow[j]) * st[j] + du * bj;
      }
    }
    if (last) {                                            // y only at final token
      float acc = 0.f;
      #pragma unroll
      for (int j = 0; j < 8; ++j)
        acc += st[j] * xdbl[15 * 48 + 20 + nh * 8 + j];    // C_T[n]
      yred[nh * DI + d] = acc;
    }
    __syncthreads();
  }

  // ================= epilogue: gate, out_proj, fc (last token only) =================
  if (tid < DI) {
    int dd = tid;
    float uu = us[15 * DI + dd];
    float y  = yred[dd] + yred[DI + dd] + uu * D_param[dd];
    float z  = xz[15 * 256 + 128 + dd];
    y *= z / (1.f + __expf(-z));                           // * silu(z)
    yvec[dd] = y;
  }
  __syncthreads();
  if (tid < DM) {
    const float* wrow = out_proj_w + tid * DI;
    float acc = 0.f;
    #pragma unroll 4
    for (int k = 0; k < DI; ++k) acc += yvec[k] * wrow[k];
    outv[tid] = acc;
  }
  __syncthreads();
  if (tid == 0) {
    float acc = fc_b[0];
    #pragma unroll 4
    for (int m = 0; m < DM; ++m) acc += outv[m] * fc_w[m];
    outp[b] = acc;
  }
}

extern "C" void kernel_launch(void* const* d_in, const int* in_sizes, int n_in,
                              void* d_out, int out_size, void* d_ws, size_t ws_size,
                              hipStream_t stream) {
  (void)in_sizes; (void)n_in; (void)out_size; (void)d_ws; (void)ws_size;
  const float* xg        = (const float*)d_in[0];
  const float* emb_w     = (const float*)d_in[1];
  const float* emb_b     = (const float*)d_in[2];
  const float* in_proj_w = (const float*)d_in[3];
  const float* conv_w    = (const float*)d_in[4];
  const float* conv_b    = (const float*)d_in[5];
  const float* x_proj_w  = (const float*)d_in[6];
  const float* dt_proj_w = (const float*)d_in[7];
  const float* dt_proj_b = (const float*)d_in[8];
  const float* A_log     = (const float*)d_in[9];
  const float* D_param   = (const float*)d_in[10];
  const float* out_proj_w= (const float*)d_in[11];
  const float* fc_w      = (const float*)d_in[12];
  const float* fc_b      = (const float*)d_in[13];
  float* outp = (float*)d_out;

  mamba_fused_kernel<<<BATCH, 256, 0, stream>>>(
      xg, emb_w, emb_b, in_proj_w, conv_w, conv_b, x_proj_w,
      dt_proj_w, dt_proj_b, A_log, D_param, out_proj_w, fc_w, fc_b, outp);
}
